// AttentionModule_89258010346004
// MI455X (gfx1250) — matrix operand order repaired and checked by reading, not compiled, not run on hardware
//
#include <hip/hip_runtime.h>

// Problem constants (from reference): B=16, IDF=128, CDF=256, H=W=128, L=64
#define B_   16
#define IDF_ 128
#define CDF_ 256
#define L_   64
#define P_   16384     // H*W
#define PT_  128       // P-tile per workgroup

typedef _Float16 half_t;
typedef __attribute__((ext_vector_type(16))) _Float16 v16h;
typedef __attribute__((ext_vector_type(8)))  _Float16 v8h;
typedef __attribute__((ext_vector_type(8)))  float    v8f;

// ---- dynamic LDS layout (byte offsets; every fragment load stays 16B-aligned) ----
#define XT_OFF     0        // half [128][136]  x-tile, f16, transposed [p][i]
#define SRC_OFF    34816    // half [128][72]   source [i][l]   (GEMM2 A)
#define SRCT_OFF   53248    // half [64][136]   source [l][i]   (GEMM1 B)
#define ATTN_OFF   70656    // float[128][68]   attn logits / softmax (f32)
#define ATTNH_OFF  105472   // half [128][72]   softmaxed attn f16 [p][l] (GEMM2 B)
#define SMEM_BYTES 123904

// Build a v16h WMMA fragment from two contiguous 8-half (16B) LDS chunks.
__device__ __forceinline__ v16h mk16(const half_t* lo, const half_t* hi) {
  v8h a = *(const v8h*)lo;
  v8h b = *(const v8h*)hi;
  v16h r;
#pragma unroll
  for (int e = 0; e < 8; ++e) { r[e] = a[e]; r[e + 8] = b[e]; }
  return r;
}

// ---------------- kernel 1: source[b] = W(128x256) @ context[b](256x64), store f16 ----------------
__global__ __launch_bounds__(256) void source_gemm_kernel(
    const float* __restrict__ ctx, const float* __restrict__ W,
    half_t* __restrict__ src) {
  const int b = blockIdx.x;
  const int t = threadIdx.x;
  const int l = t & 63;            // coalesced over l
  const int i0 = (t >> 6) * 32;    // 4 groups of 32 i-rows
  const float* ctxb = ctx + (size_t)b * CDF_ * L_;
  for (int j = 0; j < 32; ++j) {
    const int i = i0 + j;
    const float* wrow = W + (size_t)i * CDF_;
    float acc = 0.f;
#pragma unroll 8
    for (int c = 0; c < CDF_; ++c) acc = fmaf(wrow[c], ctxb[c * L_ + l], acc);
    src[((size_t)b * IDF_ + i) * L_ + l] = (half_t)acc;
  }
}

// ---------------- kernel 2: fused  attn = x^T*src -> mask+softmax -> weight = src*attn^T ----------------
__global__ __launch_bounds__(256) void fused_attn_kernel(
    const float* __restrict__ x, const half_t* __restrict__ src,
    const unsigned char* __restrict__ mask,
    float* __restrict__ weight_out, float* __restrict__ attn_out) {
  extern __shared__ char smem[];
  half_t* xT     = (half_t*)(smem + XT_OFF);
  half_t* ssrc   = (half_t*)(smem + SRC_OFF);
  half_t* ssrcT  = (half_t*)(smem + SRCT_OFF);
  float*  sattn  = (float*) (smem + ATTN_OFF);
  half_t* sattnh = (half_t*)(smem + ATTNH_OFF);

  const int b    = blockIdx.y;
  const int p0   = blockIdx.x * PT_;
  const int t    = threadIdx.x;
  const int wave = t >> 5;          // 8 wave32s
  const int lane = t & 31;
  const int hi   = lane >> 4;       // lane half-group (K selector in fragments)
  const int ln   = lane & 15;       // row/col within 16

  // ---- stage source: [i][l] and transposed [l][i] ----
  {
    const half_t* sg = src + (size_t)b * IDF_ * L_;
#pragma unroll
    for (int iter = 0; iter < 32; ++iter) {
      const int idx = iter * 256 + t;      // 8192 halfs
      const int i = idx >> 6, l = idx & 63;
      const half_t v = sg[idx];
      ssrc [i * 72  + l] = v;
      ssrcT[l * 136 + i] = v;
    }
  }

  // ---- stage x tile: fp32 -> fp16, transpose to [p][i] (coalesced float4 global reads) ----
  {
#pragma unroll
    for (int iter = 0; iter < 16; ++iter) {
      const int idx = iter * 256 + t;      // 4096 float4
      const int i = idx >> 5, p = (idx & 31) << 2;
      const float4 v = *(const float4*)(x + (((size_t)b * IDF_ + i) << 14) + p0 + p);
      xT[(p + 0) * 136 + i] = (half_t)v.x;
      xT[(p + 1) * 136 + i] = (half_t)v.y;
      xT[(p + 2) * 136 + i] = (half_t)v.z;
      xT[(p + 3) * 136 + i] = (half_t)v.w;
    }
    if (blockIdx.x + 1 < gridDim.x) {
      // prefetch part of the next P-tile (gfx1250 global_prefetch_b8)
      __builtin_prefetch(x + (((size_t)b * IDF_ + (t >> 1)) << 14) + p0 + PT_ + (t & 1) * 64, 0, 1);
    }
  }
  __syncthreads();

  // ---- GEMM1: attn[p][l] = sum_i xT[p][i]*src[i][l]   M=128(p) N=64(l) K=128(i) ----
  // Each wave owns one 16-row p-slab x all four 16-col l-tiles; A frag reused across N.
  {
    const half_t* arow = xT + (wave * 16 + ln) * 136;   // A: row m=p, K along i
    v8f acc[4] = {};
#pragma unroll
    for (int kk = 0; kk < 128; kk += 32) {
      // A 16-bit 16x32 layout: lane(hi,m) holds K = kk+hi*8+{0..7} and kk+16+hi*8+{0..7}
      const v16h a = mk16(arow + kk + hi * 8, arow + kk + 16 + hi * 8);
#pragma unroll
      for (int nt = 0; nt < 4; ++nt) {
        // B 16-bit 32x16 layout: lane(hi,n) holds K = kk+hi*8+{0..7} and kk+16+hi*8+{0..7}, n = ln
        const half_t* brow = ssrcT + (nt * 16 + ln) * 136 + kk + hi * 8;
        const v16h bb = mk16(brow, brow + 16);
        acc[nt] = __builtin_amdgcn_wmma_f32_16x16x32_f16(
            false, a, false, bb, (short)0, acc[nt], false, false);
      }
    }
    // D layout: VGPR g -> row m = g + 8*hi, col n = ln
#pragma unroll
    for (int nt = 0; nt < 4; ++nt)
#pragma unroll
      for (int g = 0; g < 8; ++g)
        sattn[(wave * 16 + hi * 8 + g) * 68 + nt * 16 + ln] = acc[nt][g];
  }
  __syncthreads();

  // ---- masked softmax over L=64 for each of the 128 p-rows (fp32, in LDS) ----
  if (t < PT_) {
    float*  row  = sattn  + t * 68;
    half_t* rowh = sattnh + t * 72;
    const unsigned char* mrow = mask + b * L_;
    float mx = -1e30f;
#pragma unroll
    for (int l = 0; l < L_; ++l) {
      const float v = mrow[l] ? -1e30f : row[l];   // mask==True -> -inf
      row[l] = v;
      mx = fmaxf(mx, v);
    }
    float s = 0.f;
#pragma unroll
    for (int l = 0; l < L_; ++l) {
      const float e = __expf(row[l] - mx);         // masked lanes underflow to 0
      s += e;
      row[l] = e;
    }
    const float inv = 1.f / s;
#pragma unroll
    for (int l = 0; l < L_; ++l) {
      const float r = row[l] * inv;
      row[l]  = r;
      rowh[l] = (half_t)r;                          // f16 copy for GEMM2 B operand
    }
  }
  __syncthreads();

  // ---- write attn_out[b][l][p0+p] (read LDS transposed -> p-contiguous coalesced stores) ----
#pragma unroll
  for (int iter = 0; iter < 32; ++iter) {
    const int idx = iter * 256 + t;                 // 8192 floats
    const int l = idx >> 7, p = idx & 127;
    attn_out[(((size_t)b * L_ + l) << 14) + p0 + p] = sattn[p * 68 + l];
  }

  // ---- GEMM2: weight[i][p] = sum_l src[i][l]*attn[p][l]   M=128(i) N=128(p) K=64(l) ----
  // Each wave owns one 16-row i-slab x all eight 16-col p-tiles; A frags (K=64) reused across N.
  {
    const half_t* arow = ssrc + (wave * 16 + ln) * 72;  // A: row m=i, K along l
    const v16h a0 = mk16(arow + hi * 8,      arow + 16 + hi * 8);
    const v16h a1 = mk16(arow + 32 + hi * 8, arow + 48 + hi * 8);
#pragma unroll
    for (int nt = 0; nt < 8; ++nt) {
      const half_t* brow = sattnh + (nt * 16 + ln) * 72; // B: [n=p][k=l]
      const v16h b0 = mk16(brow + hi * 8,      brow + 16 + hi * 8);
      const v16h b1 = mk16(brow + 32 + hi * 8, brow + 48 + hi * 8);
      v8f c = {};
      c = __builtin_amdgcn_wmma_f32_16x16x32_f16(false, a0, false, b0, (short)0, c, false, false);
      c = __builtin_amdgcn_wmma_f32_16x16x32_f16(false, a1, false, b1, (short)0, c, false, false);
#pragma unroll
      for (int g = 0; g < 8; ++g) {
        const int ig = wave * 16 + hi * 8 + g;          // D row -> i
        weight_out[(((size_t)b * IDF_ + ig) << 14) + p0 + nt * 16 + ln] = c[g];
      }
    }
  }
}

extern "C" void kernel_launch(void* const* d_in, const int* in_sizes, int n_in,
                              void* d_out, int out_size, void* d_ws, size_t ws_size,
                              hipStream_t stream) {
  (void)in_sizes; (void)n_in; (void)out_size; (void)ws_size;
  const float*         x    = (const float*)d_in[0];          // [B,IDF,H,W] f32
  const float*         ctx  = (const float*)d_in[1];          // [B,CDF,L]   f32
  const float*         W    = (const float*)d_in[2];          // [IDF,CDF]   f32
  const unsigned char* mask = (const unsigned char*)d_in[3];  // [B,L] bool (1 byte)

  float* out        = (float*)d_out;
  float* weight_out = out;                                    // B*IDF*P floats
  float* attn_out   = out + (size_t)B_ * IDF_ * P_;           // B*L*P floats
  half_t* src       = (half_t*)d_ws;                          // B*IDF*L f16 (256 KB)

  // allow >64KB dynamic LDS for the fused kernel
  (void)hipFuncSetAttribute((const void*)fused_attn_kernel,
                            hipFuncAttributeMaxDynamicSharedMemorySize, SMEM_BYTES);

  source_gemm_kernel<<<B_, 256, 0, stream>>>(ctx, W, src);
  fused_attn_kernel<<<dim3(P_ / PT_, B_), 256, SMEM_BYTES, stream>>>(
      x, src, mask, weight_out, attn_out);
}
